// PixelGatingAlignFutureRoi_29609504538963
// MI455X (gfx1250) — matrix operand order repaired and checked by reading, not compiled, not run on hardware
//
#include <hip/hip_runtime.h>
#include <hip/hip_bf16.h>

// ---------------------------------------------------------------------------
// Problem constants (from reference): x[5,B=2,C=512,H=64,W=64] fp32
// ---------------------------------------------------------------------------
#define BN 2
#define CN 512
#define HN 64
#define WN 64
#define HWN 4096           // H*W
#define FRAME_STRIDE (BN * CN * HWN)   // floats per frame of x

typedef __attribute__((ext_vector_type(16))) __bf16 v16bf;
typedef __attribute__((ext_vector_type(8)))  float  v8f;
typedef unsigned int uint32x4 __attribute__((ext_vector_type(4)));
typedef int          int32x4  __attribute__((ext_vector_type(4)));
typedef int          int32x8  __attribute__((ext_vector_type(8)));

// Tensor Data Mover availability (probe-confirmed builtins on both toolchains;
// arity differs: ROCm7.2/clang-22 = 5 args, clang-23 = 6 args)
#if defined(__has_builtin)
#if __has_builtin(__builtin_amdgcn_tensor_load_to_lds) && \
    __has_builtin(__builtin_amdgcn_s_wait_tensorcnt)
#define USE_TDM 1
#endif
#endif

// LDS row padding: 16 DWORDs of data + 4 DWORDs pad = 20-uint stride (80 B).
// Keeps every per-lane fragment chunk 16B-aligned (ds_load_b128) and gives a
// conflict-free bank spread (20*k mod 64 distinct for 16 consecutive rows).
#define WSTRIDE 20

// ----------------------------- helpers -------------------------------------
__device__ __forceinline__ unsigned short f2bf(float f) {
  unsigned u = __float_as_uint(f);
  unsigned r = u + 0x7FFFu + ((u >> 16) & 1u);   // round-to-nearest-even
  return (unsigned short)(r >> 16);
}
// order-preserving float<->uint encoding for atomic max
__device__ __forceinline__ unsigned encf(float f) {
  unsigned u = __float_as_uint(f);
  return (u & 0x80000000u) ? ~u : (u | 0x80000000u);
}
__device__ __forceinline__ float decf(unsigned k) {
  unsigned u = (k & 0x80000000u) ? (k & 0x7FFFFFFFu) : ~k;
  return __uint_as_float(u);
}

// ---------------------------------------------------------------------------
// Weight conversion: fp32 OIHW [M][OC][IC][3][3] -> bf16 [M][tap][oc][ic]
// ---------------------------------------------------------------------------
__global__ __launch_bounds__(256) void convert_w_bf16_kernel(
    const float* __restrict__ src, unsigned short* __restrict__ dst,
    int OC, int IC, long total) {
  long d = (long)blockIdx.x * blockDim.x + threadIdx.x;
  if (d >= total) return;
  int ic  = (int)(d % IC);
  long t1 = d / IC;
  int oc  = (int)(t1 % OC);
  long t2 = t1 / OC;
  int tap = (int)(t2 % 9);
  long m  = t2 / 9;
  float v = src[(((m * OC + oc) * (long)IC + ic) * 9) + tap];
  dst[d] = f2bf(v);
}

// ---------------------------------------------------------------------------
// Per-frame init: zero gate accumulator, reset global max key
// ---------------------------------------------------------------------------
__global__ __launch_bounds__(256) void frame_init_kernel(
    float* __restrict__ gwAcc, unsigned* __restrict__ maxkey, int n) {
  int i = blockIdx.x * blockDim.x + threadIdx.x;
  if (i < n) gwAcc[i] = 0.0f;
  if (i == 0) *maxkey = 0u;   // smallest possible encoded key
}

// ---------------------------------------------------------------------------
// Correlation: corr[b,d,y,x] = sum_c feat2[b,c,y,x]*feat1[b,c,y+di,x+dj]
// plus a global max (encoded-uint atomicMax).  grid = B*25*HW / 256
// ---------------------------------------------------------------------------
__global__ __launch_bounds__(256) void corr_kernel(
    const float* __restrict__ feat2, const float* __restrict__ feat1,
    float* __restrict__ corr, unsigned* __restrict__ maxkey) {
  int idx = blockIdx.x * blockDim.x + threadIdx.x;   // B*25*4096 total
  int x = idx & 63, y = (idx >> 6) & 63;
  int d = (idx >> 12) % 25;
  int b = idx / (25 * HWN);
  int ys = y + d / 5 - 2, xs = x + d % 5 - 2;
  float acc = 0.0f;
  if (ys >= 0 && ys < HN && xs >= 0 && xs < WN) {
    const float* p2 = feat2 + (size_t)b * CN * HWN + y * WN + x;
    const float* p1 = feat1 + (size_t)b * CN * HWN + ys * WN + xs;
    for (int c = 0; c < CN; ++c)
      acc = fmaf(p2[(size_t)c * HWN], p1[(size_t)c * HWN], acc);
  }
  corr[idx] = acc;
  __shared__ float sm[256];
  sm[threadIdx.x] = acc;
  __syncthreads();
  for (int s = 128; s > 0; s >>= 1) {
    if (threadIdx.x < s) sm[threadIdx.x] = fmaxf(sm[threadIdx.x], sm[threadIdx.x + s]);
    __syncthreads();
  }
  if (threadIdx.x == 0) atomicMax(maxkey, encf(sm[0]));
}

// ---------------------------------------------------------------------------
// Softmax over the 25 displacements, scaled by 20/globalmax (in place)
// ---------------------------------------------------------------------------
__global__ __launch_bounds__(256) void softmax25_kernel(
    float* __restrict__ corr, const unsigned* __restrict__ maxkey) {
  int idx = blockIdx.x * blockDim.x + threadIdx.x;   // B*4096
  int x = idx & 63, y = (idx >> 6) & 63, b = idx >> 12;
  float scale = 20.0f / decf(*maxkey);
  float v[25], m = -1e30f;
  size_t base = ((size_t)b * 25) * HWN + y * WN + x;
  for (int d = 0; d < 25; ++d) {
    v[d] = corr[base + (size_t)d * HWN] * scale;
    m = fmaxf(m, v[d]);
  }
  float s = 0.0f;
  for (int d = 0; d < 25; ++d) { v[d] = expf(v[d] - m); s += v[d]; }
  float inv = 1.0f / s;
  for (int d = 0; d < 25; ++d) corr[base + (size_t)d * HWN] = v[d] * inv;
}

// ---------------------------------------------------------------------------
// Aligned feature: new_feat[b,c,y,x] = sum_d attn[b,d,y,x]*feat1_pad shifted
// ---------------------------------------------------------------------------
__global__ __launch_bounds__(256) void align_kernel(
    const float* __restrict__ feat1, const float* __restrict__ attn,
    float* __restrict__ newf) {
  int idx = blockIdx.x * blockDim.x + threadIdx.x;   // 2^22
  int x = idx & 63, y = (idx >> 6) & 63;
  int c = (idx >> 12) & (CN - 1);
  int b = idx >> 21;
  const float* at = attn + ((size_t)b * 25) * HWN + y * WN + x;
  const float* f1 = feat1 + ((size_t)(b * CN + c)) * HWN;
  float acc = 0.0f;
  #pragma unroll
  for (int d = 0; d < 25; ++d) {
    int ys = y + d / 5 - 2, xs = x + d % 5 - 2;
    if (ys >= 0 && ys < HN && xs >= 0 && xs < WN)
      acc = fmaf(at[(size_t)d * HWN], f1[ys * WN + xs], acc);
  }
  newf[idx] = acc;
}

// ---------------------------------------------------------------------------
// Gate conv partial sums (2 out channels -> VALU with 8 channel slices)
// ---------------------------------------------------------------------------
__global__ __launch_bounds__(256) void gate_partial_kernel(
    const float* __restrict__ center, const float* __restrict__ newf,
    const float* __restrict__ Wgf /* [2][2C][3][3] of this frame */,
    float* __restrict__ gwAcc) {
  int t = blockIdx.x * blockDim.x + threadIdx.x;
  int slice = t & 7;
  int idx = t >> 3;                       // (b,g,y,x)
  int x = idx & 63, y = (idx >> 6) & 63;
  int g = (idx >> 12) & 1;
  int b = idx >> 13;
  float acc = 0.0f;
  int ic0 = slice * 128;
  for (int ic = ic0; ic < ic0 + 128; ++ic) {
    const float* src = (ic < CN)
        ? center + ((size_t)(b * CN + ic)) * HWN
        : newf + ((size_t)(b * CN + (ic - CN))) * HWN;
    const float* w = Wgf + ((size_t)g * (2 * CN) + ic) * 9;
    #pragma unroll
    for (int ky = 0; ky < 3; ++ky) {
      int ys = y + ky - 1;
      if (ys < 0 || ys >= HN) continue;
      #pragma unroll
      for (int kx = 0; kx < 3; ++kx) {
        int xs = x + kx - 1;
        if (xs < 0 || xs >= WN) continue;
        acc = fmaf(w[ky * 3 + kx], src[ys * WN + xs], acc);
      }
    }
  }
  atomicAdd(&gwAcc[idx], acc);
}

__global__ __launch_bounds__(256) void gate_sigmoid_kernel(
    float* __restrict__ gwAcc, const float* __restrict__ bgf /* [2] */) {
  int idx = blockIdx.x * blockDim.x + threadIdx.x;   // B*2*HW
  int g = (idx >> 12) & 1;
  float v = gwAcc[idx] + bgf[g];
  gwAcc[idx] = 1.0f / (1.0f + expf(-v));
}

// ---------------------------------------------------------------------------
// Gated feature, bf16: fg[b, c2, y, x] = (c2<C ? center : new_feat) * gw
// ---------------------------------------------------------------------------
__global__ __launch_bounds__(256) void fg_kernel(
    const float* __restrict__ center, const float* __restrict__ newf,
    const float* __restrict__ gw, unsigned short* __restrict__ fgBF) {
  int idx = blockIdx.x * blockDim.x + threadIdx.x;   // 2^23
  int x = idx & 63, y = (idx >> 6) & 63;
  int c2 = (idx >> 12) & (2 * CN - 1);
  int b = idx >> 22;
  int g = c2 >> 9;
  float f = g ? newf[((size_t)(b * CN + (c2 - CN))) * HWN + y * WN + x]
              : center[((size_t)(b * CN + c2)) * HWN + y * WN + x];
  float w = gw[((size_t)(b * 2 + g)) * HWN + y * WN + x];
  fgBF[idx] = f2bf(f * w);
}

// ---------------------------------------------------------------------------
// WMMA implicit-GEMM 3x3 conv (bf16 in, f32 acc).
//   block = 128 threads (4 waves), tile 64 oc x 64 px (one image row).
//   Per 32-ic chunk: stage 3-row halo input tile [3][66px][32ic] once; the
//   9 taps stream double-buffered weight tiles [64oc][32ic] via the Tensor
//   Data Mover (hardware LDS padding -> 80B rows) overlapped with compute.
//   Each wave owns a 2x2 block of 16x16 sub-tiles: 2 A-frags + 2 B-frags ->
//   4 v_wmma_f32_16x16x32_bf16 per tap, fragments via ds_load_b128.
//   grid.x = B*H (pixel rows), grid.y = OC/64.
// ---------------------------------------------------------------------------
__global__ __launch_bounds__(128) void wmma_conv3x3_kernel(
    const unsigned short* __restrict__ inBF, int IC,
    const unsigned short* __restrict__ wBF,   // [tap][oc][ic] bf16
    const float* __restrict__ bias, int OC,
    unsigned short* __restrict__ outBF, int outCtot, int chanOff,
    float* __restrict__ outF32) {
  int b = blockIdx.x >> 6;
  int y = blockIdx.x & 63;
  int ocTile = blockIdx.y * 64;

  // double-buffered weight tile [64 oc][32 ic], row stride WSTRIDE uints
  __shared__ __align__(16) unsigned int ldsW[2][64 * WSTRIDE];
  // halo input tile [3 rows][66 px][32 ic], px stride WSTRIDE uints
  __shared__ __align__(16) unsigned int ldsI[3 * 66 * WSTRIDE];
  unsigned short* ldsIs = (unsigned short*)ldsI;

  int tid = threadIdx.x;
  int wv = tid >> 5;          // wave id 0..3
  int lane = tid & 31;
  int col = lane & 15;
  int hi = lane >> 4;
  int wvOc = (wv >> 1) * 32;  // wave's oc offset within tile
  int wvPx = (wv & 1) * 32;   // wave's px offset within tile

  v8f acc[2][2];
  v8f zz = {0.f, 0.f, 0.f, 0.f, 0.f, 0.f, 0.f, 0.f};
  acc[0][0] = zz; acc[0][1] = zz; acc[1][0] = zz; acc[1][1] = zz;

#if USE_TDM
  unsigned ldsWaddr[2] = {(unsigned)(size_t)(const void*)&ldsW[0][0],
                          (unsigned)(size_t)(const void*)&ldsW[1][0]};
#endif

  // stage weight tile for `tap` into buffer `bi`
  auto stageW = [&](int tap, int bi, int ic0) {
    const unsigned short* wTap = wBF + ((size_t)tap * OC + ocTile) * IC + ic0;
#if USE_TDM
    if (wv == 0) {
      // Tensor DMA: 2D tile 32(ic) x 64(oc), 2B elems, row stride IC.
      // LDS pad: 4 DWORDs every 16 DWORDs -> 80B rows (WSTRIDE uints).
      unsigned long long ga = (unsigned long long)(size_t)wTap;
      uint32x4 g0;
      g0[0] = 0x1u;                                   // count=1, user mode
      g0[1] = ldsWaddr[bi];                           // D#.lds_addr
      g0[2] = (unsigned)(ga & 0xFFFFFFFFu);           // global_addr[31:0]
      g0[3] = (unsigned)((ga >> 32) & 0x01FFFFFFu)    // global_addr[56:32]
              | 0x80000000u;                          // type=2 (image)
      int32x8 g1;
      g1[0] = (int)((1u << 16)      // data_size = 2 bytes
                  | (1u << 20)      // pad_enable
                  | (3u << 22)      // pad_interval = 16 DWORDs
                  | (3u << 25));    // pad_amount   = 4 DWORDs
      g1[1] = (int)(((unsigned)IC & 0xFFFFu) << 16);     // tensor_dim0 lo16
      g1[2] = (int)(((unsigned)IC >> 16) | (64u << 16)); // dim0 hi | dim1 lo
      g1[3] = (int)(32u << 16);     // tensor_dim1 hi=0 | tile_dim0=32
      g1[4] = (int)64u;             // tile_dim1=64 | tile_dim2=0
      g1[5] = (int)IC;              // tensor_dim0_stride lo32
      g1[6] = 0;
      g1[7] = 0;
      int32x4 gz = {0, 0, 0, 0};
#if __clang_major__ >= 23
      int32x8 gz8 = {0, 0, 0, 0, 0, 0, 0, 0};
      __builtin_amdgcn_tensor_load_to_lds(g0, g1, gz, gz, gz8, 0);
#else
      __builtin_amdgcn_tensor_load_to_lds(g0, g1, gz, gz, 0);
#endif
    }
#else
    unsigned int* buf = &ldsW[bi][0];
    int e = tid * 8;
    #pragma unroll
    for (int r = 0; r < 2; ++r, e += 1024) {
      int oc = e >> 5;
      int icf = e & 31;
      uint4 s = *(const uint4*)(wTap + (size_t)oc * IC + icf);
      *(uint4*)&buf[oc * WSTRIDE + (icf >> 1)] = s;   // ds_store_b128
    }
#endif
  };

  for (int ic0 = 0; ic0 < IC; ic0 += 32) {
    __syncthreads();   // previous chunk done reading ldsI / both ldsW bufs
    stageW(0, 0, ic0);

    // ---- stage halo input tile once per chunk --------------------------
    {
      int ic = tid >> 2;            // 0..31
      int pxb = (tid & 3) * 16;     // 0,16,32,48
      #pragma unroll
      for (int r = 0; r < 3; ++r) {
        int ys = y + r - 1;
        union { uint4 q[2]; unsigned short s16[16]; } buf;
        if (ys >= 0 && ys < HN) {
          const unsigned short* src =
              inBF + ((size_t)(b * IC + ic0 + ic)) * HWN + ys * WN + pxb;
          buf.q[0] = *(const uint4*)(src);
          buf.q[1] = *(const uint4*)(src + 8);
        } else {
          buf.q[0] = uint4{0u, 0u, 0u, 0u};
          buf.q[1] = uint4{0u, 0u, 0u, 0u};
        }
        #pragma unroll
        for (int i = 0; i < 16; ++i)
          ldsIs[((r * 66) + pxb + 1 + i) * (2 * WSTRIDE) + ic] = buf.s16[i];
      }
      if (tid < 96) {               // zero halo columns p=0 and p=65
        int r = tid >> 5, icz = tid & 31;
        ldsIs[((r * 66) + 0) * (2 * WSTRIDE) + icz] = 0;
        ldsIs[((r * 66) + 65) * (2 * WSTRIDE) + icz] = 0;
      }
      // prefetch next chunk's input rows (global_prefetch_b8)
      if (ic0 + 32 < IC) {
        int ysn = min(max(y - 1, 0), HN - 1);
        __builtin_prefetch(
            inBF + ((size_t)(b * IC + ic0 + 32 + ic)) * HWN + ysn * WN + pxb,
            0, 0);
      }
    }

    for (int tap = 0; tap < 9; ++tap) {
      int dy = tap / 3;             // input row index 0..2
      int dx = tap % 3;             // px shift 0..2 (p = x + dx)
#if USE_TDM
      if (wv == 0) __builtin_amdgcn_s_wait_tensorcnt((unsigned short)0);
#endif
      __syncthreads();              // publish weights(tap) + input tile
      if (tap < 8) stageW(tap + 1, (tap + 1) & 1, ic0);  // overlap w/ compute

      const unsigned int* wbuf = &ldsW[tap & 1][0];
      // ---- A fragments: 2 oc sub-rows, 16B at k=8h + 16B at k=16+8h per lane ----
      union { v16bf v; uint4 q[2]; } afr[2];
      #pragma unroll
      for (int i = 0; i < 2; ++i) {
        int row = wvOc + i * 16 + col;
        int aIdx = row * WSTRIDE + hi * 4;
        afr[i].q[0] = *(const uint4*)&wbuf[aIdx];       // K 0..7  (this half)
        afr[i].q[1] = *(const uint4*)&wbuf[aIdx + 8];   // K 16..23 (this half)
      }
      // ---- 2 px sub-tiles: B fragment then 2 WMMAs each ----------------
      #pragma unroll
      for (int j = 0; j < 2; ++j) {
        union { v16bf v; uint4 q[2]; } bfr;
        int p = wvPx + j * 16 + col + dx;
        int bIdx = ((dy * 66) + p) * WSTRIDE + hi * 4;
        bfr.q[0] = *(const uint4*)&ldsI[bIdx];
        bfr.q[1] = *(const uint4*)&ldsI[bIdx + 8];
        #pragma unroll
        for (int i = 0; i < 2; ++i)
          acc[i][j] = __builtin_amdgcn_wmma_f32_16x16x32_bf16(
              false, afr[i].v, false, bfr.v, (short)0, acc[i][j], false, false);
      }
    }
  }

  __syncthreads();
  // ---- epilogue: bias + relu, store per documented C/D layout ----
  #pragma unroll
  for (int i = 0; i < 2; ++i) {
    int ocBase = ocTile + wvOc + i * 16 + hi * 8;
    #pragma unroll
    for (int j = 0; j < 2; ++j) {
      int xo = wvPx + j * 16 + col;
      #pragma unroll
      for (int g = 0; g < 8; ++g) {
        int oc = ocBase + g;
        float v = acc[i][j][g] + bias[oc];
        v = fmaxf(v, 0.0f);
        if (outF32)
          outF32[((size_t)(b * OC + oc)) * HWN + y * WN + xo] = v;
        else
          outBF[((size_t)(b * outCtot + chanOff + oc)) * HWN + y * WN + xo] =
              f2bf(v);
      }
    }
  }
}

// ---------------------------------------------------------------------------
// Host launcher
// ---------------------------------------------------------------------------
extern "C" void kernel_launch(void* const* d_in, const int* in_sizes, int n_in,
                              void* d_out, int out_size, void* d_ws,
                              size_t ws_size, hipStream_t stream) {
  const float* x  = (const float*)d_in[0];   // [5,B,C,H,W]
  const float* Wg = (const float*)d_in[1];   // [4,2,2C,3,3]
  const float* bg = (const float*)d_in[2];   // [4,2]
  const float* Wd = (const float*)d_in[3];   // [4,C,2C,3,3]
  const float* bd = (const float*)d_in[4];   // [4,C]
  const float* Wc = (const float*)d_in[5];   // [C,4C,3,3]
  const float* bc = (const float*)d_in[6];   // [C]
  float* out = (float*)d_out;                // [B,C,H,W]

  // ---- workspace carve-up (256B aligned) ----
  char* base = (char*)d_ws;
  size_t off = 0;
  auto carve = [&](size_t bytes) -> char* {
    char* p = base + off;
    off = (off + bytes + 255) & ~(size_t)255;
    return p;
  };
  float*          corr    = (float*)carve((size_t)BN * 25 * HWN * 4);
  float*          newf    = (float*)carve((size_t)BN * CN * HWN * 4);
  float*          gwAcc   = (float*)carve((size_t)BN * 2 * HWN * 4);
  unsigned*       maxkey  = (unsigned*)carve(256);
  unsigned short* fgBF    = (unsigned short*)carve((size_t)BN * 2 * CN * HWN * 2);
  unsigned short* totalBF = (unsigned short*)carve((size_t)BN * 4 * CN * HWN * 2);
  unsigned short* WdBF    = (unsigned short*)carve((size_t)4 * 9 * CN * (2 * CN) * 2);
  unsigned short* WcBF    = (unsigned short*)carve((size_t)9 * CN * (4 * CN) * 2);
  if (off > ws_size) return;   // not enough scratch; avoid OOB

  // ---- weight conversion to bf16 [m][tap][oc][ic] ----
  {
    long tWd = (long)4 * 9 * CN * (2 * CN);      // 18,874,368
    convert_w_bf16_kernel<<<(int)(tWd / 256), 256, 0, stream>>>(
        Wd, WdBF, CN, 2 * CN, tWd);
    long tWc = (long)9 * CN * (4 * CN);          // 9,437,184
    convert_w_bf16_kernel<<<(int)(tWc / 256), 256, 0, stream>>>(
        Wc, WcBF, CN, 4 * CN, tWc);
  }

  const int seq_ids[4] = {0, 1, 3, 4};
  const float* center = x + (size_t)2 * FRAME_STRIDE;

  for (int n = 0; n < 4; ++n) {
    const float* feat1 = x + (size_t)seq_ids[n] * FRAME_STRIDE;

    frame_init_kernel<<<(BN * 2 * HWN) / 256, 256, 0, stream>>>(
        gwAcc, maxkey, BN * 2 * HWN);

    corr_kernel<<<(BN * 25 * HWN) / 256, 256, 0, stream>>>(
        center, feat1, corr, maxkey);

    softmax25_kernel<<<(BN * HWN) / 256, 256, 0, stream>>>(corr, maxkey);

    align_kernel<<<(BN * CN * HWN) / 256, 256, 0, stream>>>(feat1, corr, newf);

    gate_partial_kernel<<<(BN * 2 * HWN * 8) / 256, 256, 0, stream>>>(
        center, newf, Wg + (size_t)n * 2 * (2 * CN) * 9, gwAcc);

    gate_sigmoid_kernel<<<(BN * 2 * HWN) / 256, 256, 0, stream>>>(
        gwAcc, bg + (size_t)n * 2);

    fg_kernel<<<(BN * 2 * CN * HWN) / 256, 256, 0, stream>>>(
        center, newf, gwAcc, fgBF);

    // Wd conv: bf16 WMMA implicit GEMM -> bf16 slice of totalBF
    dim3 grid(BN * HN, CN / 64);
    wmma_conv3x3_kernel<<<grid, 128, 0, stream>>>(
        fgBF, 2 * CN, WdBF + (size_t)n * 9 * CN * (2 * CN),
        bd + (size_t)n * CN, CN, totalBF, 4 * CN, n * CN, nullptr);
  }

  // final Wc conv: bf16 WMMA implicit GEMM -> fp32 d_out
  dim3 grid(BN * HN, CN / 64);
  wmma_conv3x3_kernel<<<grid, 128, 0, stream>>>(
      totalBF, 4 * CN, WcBF, bc, CN, nullptr, 0, 0, out);
}
